// SetAbstraction_76622216561210
// MI455X (gfx1250) — matrix operand order repaired and checked by reading, not compiled, not run on hardware
//
#include <hip/hip_runtime.h>
#include <hip/hip_bf16.h>

typedef _Float16 v16h __attribute__((ext_vector_type(16)));
typedef float    v8f  __attribute__((ext_vector_type(8)));

#define NB    2
#define NPTS  8192
#define NM    2048      // NPTS / STRIDE
#define KNN   32
#define C0    64
#define C1    128
#define C2    256
#define CIN   67        // 3 + C0
#define CP    96        // CIN padded to multiple of 32 (WMMA K-step)
#define R2    (0.1f * 0.1f)

// ---------------- workspace layout (bytes) ----------------
// groupA : f16 [NB*NM, 32, 96]          = 25,165,824
// W1h    : f16 [128, 96]  (zero padded) =     24,576
// W2h    : f16 [256, 128]               =     65,536
// p      : f32 [NB*NM, 256]             =  4,194,304
#define OFF_GROUP 0
#define OFF_W1H   25165824
#define OFF_W2H   25190400
#define OFF_P     25255936

// =========================================================
// 1) Furthest point sampling — one workgroup per batch.
//    2047 sequential argmax rounds; min-dist array in LDS,
//    wave-shuffle reductions, 2 barriers per round.
// =========================================================
__global__ void fps_kernel(const float* __restrict__ xyz, float* __restrict__ newxyz) {
  const int b = blockIdx.x;
  const float* P = xyz + (size_t)b * NPTS * 3;
  __shared__ float dmin[NPTS];
  __shared__ int   sel[NM];
  __shared__ float rv[32];
  __shared__ int   ri[32];
  __shared__ float lastPt[3];

  const int t = threadIdx.x, lane = t & 31, wid = t >> 5;
  for (int i = t; i < NPTS; i += 1024) dmin[i] = 1e10f;
  if (t == 0) { sel[0] = 0; lastPt[0] = P[0]; lastPt[1] = P[1]; lastPt[2] = P[2]; }
  __syncthreads();

  for (int it = 1; it < NM; ++it) {
    const float lx = lastPt[0], ly = lastPt[1], lz = lastPt[2];
    float best = -1.0f; int bi = 0x7fffffff;
    for (int i = t; i < NPTS; i += 1024) {
      float dx = P[i*3+0]-lx, dy = P[i*3+1]-ly, dz = P[i*3+2]-lz;
      float d  = dx*dx + dy*dy + dz*dz;
      float dm = dmin[i];
      dm = dm < d ? dm : d;
      dmin[i] = dm;
      if (dm > best || (dm == best && i < bi)) { best = dm; bi = i; }
    }
    // wave32 reduce (argmax, first-index tie-break)
    for (int off = 16; off > 0; off >>= 1) {
      float ov = __shfl_down(best, off, 32);
      int   oi = __shfl_down(bi,   off, 32);
      if (ov > best || (ov == best && oi < bi)) { best = ov; bi = oi; }
    }
    if (lane == 0) { rv[wid] = best; ri[wid] = bi; }
    __syncthreads();
    if (wid == 0) {
      float v2 = rv[lane]; int i2 = ri[lane];
      for (int off = 16; off > 0; off >>= 1) {
        float ov = __shfl_down(v2, off, 32);
        int   oi = __shfl_down(i2, off, 32);
        if (ov > v2 || (ov == v2 && oi < i2)) { v2 = ov; i2 = oi; }
      }
      if (lane == 0) {
        sel[it] = i2;
        lastPt[0] = P[i2*3+0]; lastPt[1] = P[i2*3+1]; lastPt[2] = P[i2*3+2];
      }
    }
    __syncthreads();
  }
  for (int m = t; m < NM; m += 1024) {
    const int ix = sel[m];
    float* o = newxyz + ((size_t)b * NM + m) * 3;
    o[0] = P[ix*3+0]; o[1] = P[ix*3+1]; o[2] = P[ix*3+2];
  }
}

// =========================================================
// 2) Weight conversion: f32 -> f16, K padded with zeros.
// =========================================================
__global__ void wconv_kernel(const float* __restrict__ W1, const float* __restrict__ W2,
                             _Float16* __restrict__ W1h, _Float16* __restrict__ W2h) {
  const int i = blockIdx.x * blockDim.x + threadIdx.x;
  if (i < C1 * CP) {
    const int n = i / CP, k = i % CP;
    W1h[i] = (k < CIN) ? (_Float16)W1[n * CIN + k] : (_Float16)0.0f;
  }
  if (i < C2 * C1) W2h[i] = (_Float16)W2[i];
}

// =========================================================
// 3) Ball query + group build — one wave per center.
//    Rank compaction via ballot; writes padded f16 A-tiles.
// =========================================================
__device__ inline void write_slot(_Float16* __restrict__ g, int cm, int slot,
                                  float rx, float ry, float rz,
                                  const float* __restrict__ F, int n) {
  _Float16* dst = g + ((size_t)cm * KNN + slot) * CP;
  dst[0] = (_Float16)rx; dst[1] = (_Float16)ry; dst[2] = (_Float16)rz;
#pragma unroll 8
  for (int c = 0; c < C0; ++c) dst[3 + c] = (_Float16)F[c * NPTS + n];
#pragma unroll
  for (int c = CIN; c < CP; ++c) dst[c] = (_Float16)0.0f;
}

__global__ void ballgroup_kernel(const float* __restrict__ xyz, const float* __restrict__ x,
                                 const float* __restrict__ newxyz, _Float16* __restrict__ groupA) {
  const int lane = threadIdx.x & 31;
  const int cm   = blockIdx.x * (blockDim.x >> 5) + (threadIdx.x >> 5);
  const int b    = cm >> 11;                 // / NM
  const float cx = newxyz[cm*3+0], cy = newxyz[cm*3+1], cz = newxyz[cm*3+2];
  const float* P = xyz + (size_t)b * NPTS * 3;
  const float* F = x   + (size_t)b * C0 * NPTS;

  int c = 0, firstIdx = -1;
  for (int chunk = 0; chunk < NPTS/32 && c < KNN; ++chunk) {
    const int n = chunk * 32 + lane;
    const float px = P[n*3+0], py = P[n*3+1], pz = P[n*3+2];
    const float dx = px-cx, dy = py-cy, dz = pz-cz;
    const bool pred = (dx*dx + dy*dy + dz*dz) < R2;
    const unsigned bits = __builtin_amdgcn_ballot_w32(pred);
    if (firstIdx < 0 && bits) firstIdx = chunk * 32 + __builtin_ctz(bits);
    const int rank = __builtin_popcount(bits & ((1u << lane) - 1u));
    const int slot = c + rank;
    if (pred && slot < KNN) write_slot(groupA, cm, slot, dx, dy, dz, F, n);
    c += __builtin_popcount(bits);
  }
  if (c < KNN) {   // pad remaining slots with the first valid point
    const int n = firstIdx;
    const float px = P[n*3+0], py = P[n*3+1], pz = P[n*3+2];
    for (int slot = c + lane; slot < KNN; slot += 32)
      write_slot(groupA, cm, slot, px-cx, py-cy, pz-cz, F, n);
  }
}

// =========================================================
// 4) Fused MLP1+ReLU -> MLP2 -> maxpool, one wave per center.
//    v_wmma_f32_16x16x32_f16 with f32 accumulation.
// =========================================================
__device__ inline v16h load_fragA(const _Float16* base, int row, int ldk, int k0, int half) {
  // A layout: lane half h holds contiguous K = k0+16h .. k0+16h+15, packed pairs
  union { v16h v; unsigned u[8]; } f;
  const unsigned* p = (const unsigned*)(base + row * ldk + k0);
  const int kb = 16 * half;
#pragma unroll
  for (int j = 0; j < 4; ++j) f.u[j]     = p[(kb >> 1) + j];
#pragma unroll
  for (int j = 0; j < 4; ++j) f.u[4 + j] = p[((kb + 8) >> 1) + j];
  return f.v;
}

__device__ inline v16h load_fragB(const _Float16* base, int n, int ldk, int k0, int half) {
  // B layout: lane half h holds contiguous K = k0+16h .. k0+16h+15 for column n
  return *(const v16h*)(base + (size_t)n * ldk + k0 + 16 * half);
}

__device__ inline v8f wmma16(v16h a, v16h b, v8f c) {
  return __builtin_amdgcn_wmma_f32_16x16x32_f16(false, a, false, b, (short)0, c, false, false);
}

#define WAVES_C 4
__global__ void mlp_kernel(const _Float16* __restrict__ groupA,
                           const _Float16* __restrict__ W1h, const float* __restrict__ b1,
                           const _Float16* __restrict__ W2h, const float* __restrict__ b2,
                           float* __restrict__ p) {
  __shared__ _Float16 hbuf[WAVES_C][KNN * C1];   // 4 * 8KB
  const int wid  = threadIdx.x >> 5;
  const int lane = threadIdx.x & 31;
  const int cm   = blockIdx.x * WAVES_C + wid;
  const int half = lane >> 4, r16 = lane & 15;
  const _Float16* Ag = groupA + (size_t)cm * KNN * CP;
  _Float16* hw = hbuf[wid];

  // ---- layer 1: (32x96) @ (96x128)^T, relu, -> LDS f16 ----
  v16h a1[2][3];
#pragma unroll
  for (int rt = 0; rt < 2; ++rt)
#pragma unroll
    for (int kk = 0; kk < 3; ++kk)
      a1[rt][kk] = load_fragA(Ag, rt * 16 + r16, CP, kk * 32, half);

#pragma unroll
  for (int ct = 0; ct < C1 / 16; ++ct) {
    v8f acc0 = {0,0,0,0,0,0,0,0}, acc1 = {0,0,0,0,0,0,0,0};
    const int col = ct * 16 + r16;
    const float bb = b1[col];
#pragma unroll
    for (int kk = 0; kk < 3; ++kk) {
      v16h bf = load_fragB(W1h, col, CP, kk * 32, half);
      acc0 = wmma16(a1[0][kk], bf, acc0);
      acc1 = wmma16(a1[1][kk], bf, acc1);
    }
#pragma unroll
    for (int i = 0; i < 8; ++i) {
      float v0 = acc0[i] + bb; v0 = v0 > 0.0f ? v0 : 0.0f;
      float v1 = acc1[i] + bb; v1 = v1 > 0.0f ? v1 : 0.0f;
      hw[(i + 8 * half) * C1 + col]      = (_Float16)v0;   // rows 0..15
      hw[(16 + i + 8 * half) * C1 + col] = (_Float16)v1;   // rows 16..31
    }
  }
  __syncthreads();

  // ---- layer 2: (32x128) @ (128x256)^T, then max over 32 rows ----
  v16h a2[2][4];
#pragma unroll
  for (int rt = 0; rt < 2; ++rt)
#pragma unroll
    for (int kk = 0; kk < 4; ++kk)
      a2[rt][kk] = load_fragA(hw, rt * 16 + r16, C1, kk * 32, half);

#pragma unroll
  for (int ct = 0; ct < C2 / 16; ++ct) {
    v8f acc0 = {0,0,0,0,0,0,0,0}, acc1 = {0,0,0,0,0,0,0,0};
    const int col = ct * 16 + r16;
#pragma unroll
    for (int kk = 0; kk < 4; ++kk) {
      v16h bf = load_fragB(W2h, col, C1, kk * 32, half);
      acc0 = wmma16(a2[0][kk], bf, acc0);
      acc1 = wmma16(a2[1][kk], bf, acc1);
    }
    float m = -3.0e38f;
#pragma unroll
    for (int i = 0; i < 8; ++i) { m = fmaxf(m, acc0[i]); m = fmaxf(m, acc1[i]); }
    m = fmaxf(m, __shfl_xor(m, 16, 32));   // combine row halves (same column)
    if (half == 0) p[(size_t)cm * C2 + col] = m + b2[col];
  }
}

// =========================================================
// 5) BatchNorm over (B,M) per channel + transposed store.
// =========================================================
__global__ void bn_kernel(const float* __restrict__ p, const float* __restrict__ gamma,
                          const float* __restrict__ beta, float* __restrict__ out) {
  const int ch = blockIdx.x, t = threadIdx.x;
  __shared__ float ssum[256], ssq[256];
  float s = 0.0f, q = 0.0f;
  for (int i = t; i < NB * NM; i += 256) {
    const float v = p[(size_t)i * C2 + ch];
    s += v; q += v * v;
  }
  ssum[t] = s; ssq[t] = q;
  __syncthreads();
  for (int off = 128; off > 0; off >>= 1) {
    if (t < off) { ssum[t] += ssum[t + off]; ssq[t] += ssq[t + off]; }
    __syncthreads();
  }
  const float inv_n = 1.0f / (float)(NB * NM);
  const float mean  = ssum[0] * inv_n;
  const float var   = ssq[0] * inv_n - mean * mean;
  const float scale = gamma[ch] * rsqrtf(var + 1e-5f);
  const float shift = beta[ch] - mean * scale;
  for (int i = t; i < NB * NM; i += 256) {
    const int b = i >> 11, m = i & (NM - 1);
    out[((size_t)b * C2 + ch) * NM + m] = p[(size_t)i * C2 + ch] * scale + shift;
  }
}

// =========================================================
extern "C" void kernel_launch(void* const* d_in, const int* in_sizes, int n_in,
                              void* d_out, int out_size, void* d_ws, size_t ws_size,
                              hipStream_t stream) {
  const float* xyz   = (const float*)d_in[0];
  const float* x     = (const float*)d_in[1];
  const float* W1    = (const float*)d_in[2];
  const float* b1    = (const float*)d_in[3];
  const float* W2    = (const float*)d_in[4];
  const float* b2    = (const float*)d_in[5];
  const float* gamma = (const float*)d_in[6];
  const float* beta  = (const float*)d_in[7];

  float* out    = (float*)d_out;
  float* newxyz = out;                       // (B, M, 3) = 12288 floats
  float* outbn  = out + (size_t)NB * NM * 3; // (B, C2, M)

  char* ws = (char*)d_ws;
  _Float16* groupA = (_Float16*)(ws + OFF_GROUP);
  _Float16* W1h    = (_Float16*)(ws + OFF_W1H);
  _Float16* W2h    = (_Float16*)(ws + OFF_W2H);
  float*    p      = (float*)   (ws + OFF_P);

  fps_kernel      <<<NB, 1024, 0, stream>>>(xyz, newxyz);
  wconv_kernel    <<<(C2 * C1 + 255) / 256, 256, 0, stream>>>(W1, W2, W1h, W2h);
  ballgroup_kernel<<<(NB * NM) / 8, 256, 0, stream>>>(xyz, x, newxyz, groupA);
  mlp_kernel      <<<(NB * NM) / WAVES_C, WAVES_C * 32, 0, stream>>>(groupA, W1h, b1, W2h, b2, p);
  bn_kernel       <<<C2, 256, 0, stream>>>(p, gamma, beta, outbn);
}
